// RWKVTimeMixing_6803228197622
// MI455X (gfx1250) — matrix operand order repaired and checked by reading, not compiled, not run on hardware
//
#include <hip/hip_runtime.h>
#include <hip/hip_bf16.h>
#include <math.h>

// ---------------- problem constants ----------------
#define BB 8
#define TT 4096
#define DD 1024
#define SS 64
#define MM (BB*TT)          // 32768 rows
#define CHUNK 64            // recurrence chunk length
#define NCHUNK (TT/CHUNK)   // 64 chunks
#define ROWS 32             // M-tile per block for the big GEMMs

typedef __attribute__((ext_vector_type(16))) __bf16 bf16x16;
typedef __attribute__((ext_vector_type(8)))  float  f32x8;

union FragB { bf16x16 v; unsigned int d[8]; };

__device__ __forceinline__ unsigned short f2bf(float f) {
    unsigned int x = __float_as_uint(f);
    unsigned int r = (x + 0x7fffu + ((x >> 16) & 1u)) >> 16;   // RNE
    return (unsigned short)r;
}

// A fragment (16x32 bf16, M x K), contiguous k order used by this code:
//  lanes 0-15: M=lane,  K = 0..15; lanes 16-31: M=lane-16, K = 16..31
__device__ __forceinline__ bf16x16 load_frag_a(const unsigned short* row, int lane) {
    FragB f;
    const unsigned short* p = row + ((lane & 16) ? 16 : 0);
    ((uint4*)f.d)[0] = *(const uint4*)(p);
    ((uint4*)f.d)[1] = *(const uint4*)(p + 8);
    return f.v;
}

// A fragment from a 32-row x 32-col LDS tile (row stride 32 elements)
__device__ __forceinline__ bf16x16 frag_a_lds(const unsigned short* ab, int rg, int lane) {
    FragB f;
    const unsigned short* p = ab + (rg * 16 + (lane & 15)) * 32 + ((lane & 16) ? 16 : 0);
    ((uint4*)f.d)[0] = *(const uint4*)(p);
    ((uint4*)f.d)[1] = *(const uint4*)(p + 8);
    return f.v;
}

// B fragment (32x16 bf16, K x N): lanes 0-15: N=lane, K=k0..k0+15;
// lanes 16-31: N=lane-16, K=k0+16..k0+31. B[k,n] = W[n,k] row-major ->
// 16 contiguous bf16 per lane (32B, 16B-aligned).
__device__ __forceinline__ bf16x16 load_frag_b(const unsigned short* p) {
    FragB f;
    const uint4* q = (const uint4*)p;
    ((uint4*)f.d)[0] = q[0];
    ((uint4*)f.d)[1] = q[1];
    return f.v;
}

// B fragment from an LDS tile laid out [n][32] (row stride 32 elements)
__device__ __forceinline__ bf16x16 frag_b_lds(const unsigned short* bb, int n, int lane) {
    FragB f;
    const unsigned short* p = bb + n * 32 + ((lane & 16) ? 16 : 0);
    ((uint4*)f.d)[0] = *(const uint4*)(p);
    ((uint4*)f.d)[1] = *(const uint4*)(p + 8);
    return f.v;
}

__device__ __forceinline__ f32x8 wmma_bf16(bf16x16 a, bf16x16 b, f32x8 c) {
    return __builtin_amdgcn_wmma_f32_16x16x32_bf16(false, a, false, b, (short)0, c, false, false);
}

// CDNA5 async global->LDS copy (ASYNCcnt-tracked), 16B per lane.
__device__ __forceinline__ void async_copy_b128(unsigned lds_off, const void* gptr) {
    unsigned long long ga = (unsigned long long)gptr;
    asm volatile("global_load_async_to_lds_b128 %0, %1, off"
                 :: "v"(lds_off), "v"(ga) : "memory");
}
__device__ __forceinline__ void wait_async0() {
    asm volatile("s_wait_asynccnt 0x0" ::: "memory");
}

// ---------------- K0: conversions ----------------
__global__ void k_f2bf(const float* __restrict__ src, unsigned short* __restrict__ dst, int n) {
    for (int i = blockIdx.x * blockDim.x + threadIdx.x; i < n; i += gridDim.x * blockDim.x)
        dst[i] = f2bf(src[i]);
}

__global__ void k_gate(const float* __restrict__ sg, float* __restrict__ gate, int n) {
    int i = blockIdx.x * blockDim.x + threadIdx.x;
    if (i < n) gate[i] = 1.0f / (1.0f + __expf(-sg[i]));
}

// ---------------- K1: shift GEMM + blend + LayerNorm (fused) ----------------
// Block: 32 rows x 1024 cols. 8 waves, each 32x128 (16 WMMA tiles).
// BOTH operand tiles staged in LDS via CDNA5 async copies, double-buffered:
//   A: 32x32 bf16 (2KB/buf), B: 1024x32 bf16 (64KB/buf). Total LDS ~134KB.
// LayerNorm is register-resident: shfl partials + 2KB LDS exchange.
__global__ __launch_bounds__(256) void k_shift_ln(
    const unsigned short* __restrict__ xb,    // x bf16 (M,D)
    const float*          __restrict__ x,     // original x fp32 (M,D)
    const unsigned short* __restrict__ Wsh,   // W_shift bf16 (D,D) row-major
    const float*          __restrict__ gate,  // sigmoid(shift_gate) (D)
    const float*          __restrict__ ln_w,
    const float*          __restrict__ ln_b,
    unsigned short*       __restrict__ xn)    // out: normalized, bf16 (M,D)
{
    __shared__ unsigned short Abuf[2][ROWS * 32];   // 4 KB
    __shared__ unsigned short Bbuf[2][DD * 32];     // 128 KB
    __shared__ float part[ROWS][8][2];              // 2 KB

    const int tid  = threadIdx.x;
    const int lane = tid & 31;
    const int wv   = tid >> 5;
    const int m0   = blockIdx.x * ROWS;

    // shifted A rows: x_cat[b,t] = x[b, (t+T/2) % T]; 32 aligned rows don't wrap
    const int b   = m0 >> 12;
    const int ts0 = ((m0 & (TT - 1)) + TT / 2) & (TT - 1);
    const unsigned short* Aglob = xb + (((size_t)(b << 12) + ts0) << 10);

    const unsigned abase[2] = { (unsigned)(size_t)&Abuf[0][0],
                                (unsigned)(size_t)&Abuf[1][0] };
    const unsigned bbase[2] = { (unsigned)(size_t)&Bbuf[0][0],
                                (unsigned)(size_t)&Bbuf[1][0] };
    const int cr = tid >> 1, ch = tid & 1;   // A-copy row / 16B half (threads 0..127)

    f32x8 acc[2][8] = {};
    const int bsel = (lane & 16) ? 16 : 0;   // (unused for LDS B frag; kept for clarity)
    const int nl   = lane & 15;
    const int g    = lane >> 4;
    (void)bsel;

    // issue async copies for k-step kt into buffer bufi
    auto issue = [&](int kt, int bufi) {
        const int k0 = kt * 32;
        if (tid < 128)
            async_copy_b128(abase[bufi] + (unsigned)(cr * 64 + ch * 32),
                            Aglob + (size_t)cr * DD + k0 + ch * 16);
        // B tile: 1024 rows x 32 cols bf16 = 4096 x 16B chunks
#pragma unroll
        for (int i = 0; i < 16; ++i) {
            int c = tid + i * 256;
            int n = c >> 2, p = c & 3;
            async_copy_b128(bbase[bufi] + (unsigned)(n * 64 + p * 16),
                            Wsh + (size_t)n * DD + k0 + p * 8);
        }
    };

    issue(0, 0);
    for (int kt = 0; kt < DD / 32; ++kt) {
        wait_async0();
        __syncthreads();
        if (kt + 1 < DD / 32) issue(kt + 1, (kt + 1) & 1);

        const unsigned short* ab = Abuf[kt & 1];
        const unsigned short* bb = Bbuf[kt & 1];
        bf16x16 a0 = frag_a_lds(ab, 0, lane);
        bf16x16 a1 = frag_a_lds(ab, 1, lane);
#pragma unroll
        for (int j = 0; j < 8; ++j) {
            int n = (wv * 8 + j) * 16 + nl;
            bf16x16 bf = frag_b_lds(bb, n, lane);
            acc[0][j] = wmma_bf16(a0, bf, acc[0][j]);
            acc[1][j] = wmma_bf16(a1, bf, acc[1][j]);
        }
    }

    // ---- gated blend (in accumulators) + per-row partial sums ----
    float gt[8], lnw[8], lnb[8];
#pragma unroll
    for (int j = 0; j < 8; ++j) {
        int n = (wv * 8 + j) * 16 + nl;
        gt[j] = gate[n]; lnw[j] = ln_w[n]; lnb[j] = ln_b[n];
    }
    float rs[2][8], rq[2][8];
#pragma unroll
    for (int rg = 0; rg < 2; ++rg)
#pragma unroll
        for (int r = 0; r < 8; ++r) { rs[rg][r] = 0.f; rq[rg][r] = 0.f; }

#pragma unroll
    for (int rg = 0; rg < 2; ++rg)
#pragma unroll
        for (int j = 0; j < 8; ++j) {
            int n = (wv * 8 + j) * 16 + nl;
#pragma unroll
            for (int r = 0; r < 8; ++r) {
                int m = m0 + rg * 16 + 8 * g + r;
                float y = acc[rg][j][r] * gt[j] + x[(size_t)m * DD + n] * (1.0f - gt[j]);
                acc[rg][j][r] = y;
                rs[rg][r] += y;
                rq[rg][r] += y * y;
            }
        }
#pragma unroll
    for (int o = 1; o < 16; o <<= 1)
#pragma unroll
        for (int rg = 0; rg < 2; ++rg)
#pragma unroll
            for (int r = 0; r < 8; ++r) {
                rs[rg][r] += __shfl_xor(rs[rg][r], o, 32);
                rq[rg][r] += __shfl_xor(rq[rg][r], o, 32);
            }
    if (nl == 0) {
#pragma unroll
        for (int rg = 0; rg < 2; ++rg)
#pragma unroll
            for (int r = 0; r < 8; ++r) {
                int row = rg * 16 + 8 * g + r;
                part[row][wv][0] = rs[rg][r];
                part[row][wv][1] = rq[rg][r];
            }
    }
    __syncthreads();

    // ---- finish LN and emit bf16 ----
#pragma unroll
    for (int rg = 0; rg < 2; ++rg)
#pragma unroll
        for (int r = 0; r < 8; ++r) {
            int row = rg * 16 + 8 * g + r;
            float s = 0.f, q = 0.f;
#pragma unroll
            for (int w2 = 0; w2 < 8; ++w2) { s += part[row][w2][0]; q += part[row][w2][1]; }
            float mu  = s * (1.0f / DD);
            float var = q * (1.0f / DD) - mu * mu;
            float inv = rsqrtf(var + 1e-5f);
            size_t base = (size_t)(m0 + row) * DD;
#pragma unroll
            for (int j = 0; j < 8; ++j) {
                int n = (wv * 8 + j) * 16 + nl;
                float v = (acc[rg][j][r] - mu) * inv * lnw[j] + lnb[j];
                xn[base + n] = f2bf(v);
            }
        }
}

// ---------------- K2: k/v projections + wkv (fused) ----------------
__global__ __launch_bounds__(256) void k_kv_wkv(
    const unsigned short* __restrict__ xn,
    const unsigned short* __restrict__ Wk,   // (S,D) bf16
    const unsigned short* __restrict__ Wv,   // (S,D) bf16
    const float*          __restrict__ time_first,
    float*                __restrict__ wkv)  // (M,S) fp32
{
    const int lane  = threadIdx.x & 31;
    const int wv_id = threadIdx.x >> 5;
    const int m0    = blockIdx.x * 32 + (wv_id >> 2) * 16;
    const int ntile = wv_id & 3;

    const unsigned short* arow = xn + (size_t)(m0 + (lane & 15)) * DD;
    const int nrow = ntile * 16 + (lane & 15);
    const int bsel = (lane & 16) ? 16 : 0;

    f32x8 ka = {}, va = {};
    for (int k0 = 0; k0 < DD; k0 += 32) {
        bf16x16 a  = load_frag_a(arow + k0, lane);
        bf16x16 bk = load_frag_b(Wk + (size_t)nrow * DD + k0 + bsel);
        bf16x16 bv = load_frag_b(Wv + (size_t)nrow * DD + k0 + bsel);
        ka = wmma_bf16(a, bk, ka);
        va = wmma_bf16(a, bv, va);
    }

    const int g = lane >> 4;
    const int s = ntile * 16 + (lane & 15);
    float tf = __expf(time_first[s]);
#pragma unroll
    for (int r = 0; r < 8; ++r) {
        int m = m0 + 8 * g + r;
        float sg = 1.0f / (1.0f + __expf(-ka[r]));
        wkv[(size_t)m * SS + s] = __expf(-tf * sg) * va[r];
    }
}

// ---------------- K3: chunked linear-recurrence scan ----------------
__global__ void k_scan_chunk(const float* __restrict__ time_decay,
                             float* __restrict__ st, float* __restrict__ Ebuf)
{
    int tid = blockIdx.x * blockDim.x + threadIdx.x;
    if (tid >= BB * NCHUNK * SS) return;
    int s = tid & (SS - 1);
    int c = (tid >> 6) & (NCHUNK - 1);
    int b = tid >> 12;
    float w = __expf(time_decay[s]);
    float p = 0.f;
    size_t base = ((size_t)b * TT + (size_t)c * CHUNK) * SS + s;
    for (int j = 0; j < CHUNK; ++j) {
        size_t idx = base + (size_t)j * SS;
        p = p * w + st[idx];
        st[idx] = p;
    }
    Ebuf[tid] = p;
}

__global__ void k_scan_carry(const float* __restrict__ time_decay,
                             const float* __restrict__ Ebuf,
                             float* __restrict__ carryIn, float* __restrict__ last_out)
{
    int tid = blockIdx.x * blockDim.x + threadIdx.x;
    if (tid >= BB * SS) return;
    int s = tid & (SS - 1);
    int b = tid >> 6;
    float wL = __expf((float)CHUNK * time_decay[s]);
    float carry = 0.f;
    for (int c = 0; c < NCHUNK; ++c) {
        size_t i = ((size_t)b * NCHUNK + c) * SS + s;
        carryIn[i] = carry;
        carry = Ebuf[i] + wL * carry;
    }
    last_out[(size_t)b * SS + s] = carry;
}

__global__ void k_apply_carry(const float* __restrict__ time_decay,
                              const float* __restrict__ st,
                              const float* __restrict__ carryIn,
                              unsigned short* __restrict__ stb)
{
    size_t tid = (size_t)blockIdx.x * blockDim.x + threadIdx.x;
    if (tid >= (size_t)MM * SS) return;
    int s = (int)(tid & (SS - 1));
    size_t m = tid >> 6;
    int t = (int)(m & (TT - 1));
    int b = (int)(m >> 12);
    int c = t >> 6, j = t & (CHUNK - 1);
    float wp = __expf((float)(j + 1) * time_decay[s]);
    float v = st[tid] + wp * carryIn[((size_t)b * NCHUNK + c) * SS + s];
    stb[tid] = f2bf(v);
}

// ---------------- K4: output GEMM (32 rows x 1024 cols, K = 64) ----------------
__global__ __launch_bounds__(256) void k_out(
    const unsigned short* __restrict__ stb,  // (M,S) bf16
    const unsigned short* __restrict__ Wo,   // (D,S) bf16
    float*                __restrict__ out)  // (M,D) fp32
{
    const int lane = threadIdx.x & 31;
    const int wv   = threadIdx.x >> 5;
    const int m0   = blockIdx.x * ROWS;
    const unsigned short* arow0 = stb + (size_t)(m0 + (lane & 15)) * SS;
    const unsigned short* arow1 = arow0 + 16 * SS;
    const int bsel = (lane & 16) ? 16 : 0;

    f32x8 acc[2][8] = {};
    for (int k0 = 0; k0 < SS; k0 += 32) {
        bf16x16 a0 = load_frag_a(arow0 + k0, lane);
        bf16x16 a1 = load_frag_a(arow1 + k0, lane);
#pragma unroll
        for (int j = 0; j < 8; ++j) {
            int n = (wv * 8 + j) * 16 + (lane & 15);
            bf16x16 bf = load_frag_b(Wo + (size_t)n * SS + k0 + bsel);
            acc[0][j] = wmma_bf16(a0, bf, acc[0][j]);
            acc[1][j] = wmma_bf16(a1, bf, acc[1][j]);
        }
    }
    const int g  = lane >> 4;
    const int nl = lane & 15;
#pragma unroll
    for (int rg = 0; rg < 2; ++rg)
#pragma unroll
        for (int j = 0; j < 8; ++j) {
            int n = (wv * 8 + j) * 16 + nl;
#pragma unroll
            for (int r = 0; r < 8; ++r)
                out[(size_t)(m0 + rg * 16 + 8 * g + r) * DD + n] = acc[rg][j][r];
        }
}

// ---------------- launcher ----------------
extern "C" void kernel_launch(void* const* d_in, const int* in_sizes, int n_in,
                              void* d_out, int out_size, void* d_ws, size_t ws_size,
                              hipStream_t stream) {
    const float* x          = (const float*)d_in[0];
    const float* time_decay = (const float*)d_in[1];
    const float* time_first = (const float*)d_in[2];
    const float* W_key      = (const float*)d_in[3];
    const float* W_value    = (const float*)d_in[4];
    const float* W_output   = (const float*)d_in[5];
    const float* W_shift    = (const float*)d_in[6];
    const float* shift_gate = (const float*)d_in[7];
    const float* ln_w       = (const float*)d_in[8];
    const float* ln_b       = (const float*)d_in[9];

    float* out        = (float*)d_out;                    // (B,T,D)
    float* last_state = out + (size_t)MM * DD;            // (B,S)

    // workspace carve-up
    char* w = (char*)d_ws;
    unsigned short* xb   = (unsigned short*)w; w += (size_t)MM * DD * 2;   // 64 MB
    unsigned short* xn   = (unsigned short*)w; w += (size_t)MM * DD * 2;   // 64 MB
    unsigned short* Wshb = (unsigned short*)w; w += (size_t)DD * DD * 2;   // 2 MB
    unsigned short* Wkb  = (unsigned short*)w; w += (size_t)SS * DD * 2;
    unsigned short* Wvb  = (unsigned short*)w; w += (size_t)SS * DD * 2;
    unsigned short* Wob  = (unsigned short*)w; w += (size_t)DD * SS * 2;
    float* gate          = (float*)w;          w += (size_t)DD * 4;
    float* wkv           = (float*)w;          w += (size_t)MM * SS * 4;   // reused as states
    float* Ebuf          = (float*)w;          w += (size_t)BB * NCHUNK * SS * 4;
    float* carryIn       = (float*)w;          w += (size_t)BB * NCHUNK * SS * 4;
    unsigned short* stb  = (unsigned short*)w; w += (size_t)MM * SS * 2;

    // K0: conversions
    k_f2bf<<<4096, 256, 0, stream>>>(x, xb, MM * DD);
    k_f2bf<<<1024, 256, 0, stream>>>(W_shift, Wshb, DD * DD);
    k_f2bf<<<256,  256, 0, stream>>>(W_key,   Wkb, SS * DD);
    k_f2bf<<<256,  256, 0, stream>>>(W_value, Wvb, SS * DD);
    k_f2bf<<<256,  256, 0, stream>>>(W_output, Wob, DD * SS);
    k_gate<<<4,    256, 0, stream>>>(shift_gate, gate, DD);

    // K1: shift GEMM + blend + LN (A and B async-LDS staged, double buffered)
    k_shift_ln<<<MM / ROWS, 256, 0, stream>>>(xb, x, Wshb, gate, ln_w, ln_b, xn);

    // K2: k/v + wkv
    k_kv_wkv<<<MM / 32, 256, 0, stream>>>(xn, Wkb, Wvb, time_first, wkv);

    // K3: chunked scan
    k_scan_chunk<<<(BB * NCHUNK * SS + 255) / 256, 256, 0, stream>>>(time_decay, wkv, Ebuf);
    k_scan_carry<<<(BB * SS + 255) / 256, 256, 0, stream>>>(time_decay, Ebuf, carryIn, last_state);
    k_apply_carry<<<(int)(((size_t)MM * SS + 255) / 256), 256, 0, stream>>>(time_decay, wkv, carryIn, stb);

    // K4: output GEMM
    k_out<<<MM / ROWS, 256, 0, stream>>>(stb, Wob, out);
}
